// Mamba3Block_91328184582833
// MI455X (gfx1250) — compile-verified, matrix operand order checked by reading, not run on hardware
//
#include <hip/hip_runtime.h>
#include <math.h>

typedef __attribute__((ext_vector_type(16))) __bf16 bf16x16;
typedef __attribute__((ext_vector_type(8)))  __bf16 bf16x8;
typedef __attribute__((ext_vector_type(8)))  float  f32x8;
typedef int v4i __attribute__((vector_size(16)));

#define DEV static __device__ __forceinline__

// ---------- CDNA5 async global->LDS path (guarded) ----------
#if defined(__HIP_DEVICE_COMPILE__) && __has_builtin(__builtin_amdgcn_global_load_async_to_lds_b128)
#define HAS_ASYNC_LDS 1
#else
#define HAS_ASYNC_LDS 0
#endif

#if HAS_ASYNC_LDS
DEV void async_b128(const void* g, void* l) {
    __builtin_amdgcn_global_load_async_to_lds_b128(
        (__attribute__((address_space(1))) v4i*)(unsigned long long)(g),
        (__attribute__((address_space(3))) v4i*)(unsigned long long)(l),
        0, 0);
}
#if __has_builtin(__builtin_amdgcn_s_wait_asynccnt)
#define WAIT_ASYNC(n) __builtin_amdgcn_s_wait_asynccnt(n)
#else
#define WAIT_ASYNC(n) asm volatile("s_wait_asynccnt %0" ::"i"(n) : "memory")
#endif
#else
#define WAIT_ASYNC(n) ((void)0)
#endif

// ---------- helpers ----------
DEV unsigned short f2bf(float f) {
    unsigned int x = __float_as_uint(f);
    x += 0x7fffu + ((x >> 16) & 1u);   // round to nearest even
    return (unsigned short)(x >> 16);
}
DEV float sigm(float x)      { return 1.0f / (1.0f + __expf(-x)); }
DEV float siluf(float x)     { return x * sigm(x); }
DEV float softplusf(float x) { return x > 20.0f ? x : log1pf(__expf(x)); }

// ---------- sizes ----------
#define BATCH   4
#define SEQ     4096
#define ROWS    16384        // BATCH*SEQ
#define DM      1024
#define DI      2048
#define NH      16
#define HD      128
#define DS      64
#define PROJN   4256
#define NPAD    4352         // 34*128

// ---------- weight convert+transpose (f32 [K x N] -> bf16 [N x K], zero-pad rows) ----------
__global__ __launch_bounds__(256) void k_cvt_t(const float* __restrict__ src,
                                               unsigned short* __restrict__ dst,
                                               int K, int srcN, long long total) {
    long long i = (long long)blockIdx.x * 256 + threadIdx.x;
    if (i >= total) return;
    long long n = i / K;
    int k = (int)(i - n * K);
    dst[i] = (n < srcN) ? f2bf(src[(size_t)k * srcN + n]) : (unsigned short)0;
}

// ---------- layernorm -> bf16 ----------
__global__ __launch_bounds__(256) void k_layernorm(const float* __restrict__ u,
                                                   const float* __restrict__ sc,
                                                   const float* __restrict__ bi,
                                                   unsigned short* __restrict__ xn) {
    int row = blockIdx.x;
    const float* ur = u + (size_t)row * DM;
    __shared__ float red[256];
    float v[4]; float s = 0.f;
    #pragma unroll
    for (int i = 0; i < 4; i++) { v[i] = ur[threadIdx.x + 256 * i]; s += v[i]; }
    red[threadIdx.x] = s; __syncthreads();
    for (int o = 128; o > 0; o >>= 1) { if (threadIdx.x < o) red[threadIdx.x] += red[threadIdx.x + o]; __syncthreads(); }
    float mu = red[0] * (1.0f / DM); __syncthreads();
    float ss = 0.f;
    #pragma unroll
    for (int i = 0; i < 4; i++) { float d = v[i] - mu; ss += d * d; }
    red[threadIdx.x] = ss; __syncthreads();
    for (int o = 128; o > 0; o >>= 1) { if (threadIdx.x < o) red[threadIdx.x] += red[threadIdx.x + o]; __syncthreads(); }
    float inv = rsqrtf(red[0] * (1.0f / DM) + 1e-6f);
    #pragma unroll
    for (int i = 0; i < 4; i++) {
        int c = threadIdx.x + 256 * i;
        xn[(size_t)row * DM + c] = f2bf((v[i] - mu) * inv * sc[c] + bi[c]);
    }
}

// ---------- theta = u @ theta_w  (16384 x 1024 x 32) ----------
__global__ __launch_bounds__(256) void k_theta(const float* __restrict__ u,
                                               const float* __restrict__ tw,
                                               float* __restrict__ theta) {
    int lr = threadIdx.x >> 5;             // 8 rows / block
    int j  = threadIdx.x & 31;             // 32 cols
    int row = blockIdx.x * 8 + lr;
    const float* ur = u + (size_t)row * DM;
    float acc = 0.f;
    for (int k = 0; k < DM; k++) acc = fmaf(ur[k], tw[k * 32 + j], acc);
    theta[(size_t)row * 32 + j] = acc;
}

// ---------- bf16 WMMA GEMM: C[MxN] = A[MxK] * Bt[NxK]^T (+residual) ----------
// block tile 128x128, BK=32; 8 waves: wave grid 4(m) x 2(n), each wave 32x64 = 2x4 frags.
// Double-buffered LDS; async global->LDS fills when available.
template <int EPI>
__global__ __launch_bounds__(256) void k_gemm(const unsigned short* __restrict__ A,
                                              const unsigned short* __restrict__ Bt,
                                              float* __restrict__ C,
                                              const float* __restrict__ resid,
                                              int M, int N, int K) {
    __shared__ __align__(16) unsigned short sA[2][128][40];   // [m][k], padded rows (80B)
    __shared__ __align__(16) unsigned short sB[2][128][40];   // [n][k], padded rows
    const int tid = threadIdx.x;
    const int bm0 = blockIdx.y * 128, bn0 = blockIdx.x * 128;
    const int w = tid >> 5, lane = tid & 31;
    const int wm = w >> 1, wn = w & 1;
    const int lrow = lane & 15, lhalf = lane >> 4;

    f32x8 acc[2][4];
    #pragma unroll
    for (int tm = 0; tm < 2; tm++)
        #pragma unroll
        for (int tn = 0; tn < 4; tn++)
            #pragma unroll
            for (int r = 0; r < 8; r++) acc[tm][tn][r] = 0.0f;

    auto load_tile = [&](int buf, int kt) {
        #pragma unroll
        for (int it = 0; it < 2; it++) {
            int c = tid + it * 256;              // 0..511
            int r = c >> 2, ko = (c & 3) * 8;    // 128 rows x 4 chunks of 8 bf16
            const unsigned short* ga = A  + (size_t)(bm0 + r) * K + kt + ko;
            const unsigned short* gb = Bt + (size_t)(bn0 + r) * K + kt + ko;
#if HAS_ASYNC_LDS
            async_b128(ga, &sA[buf][r][ko]);
            async_b128(gb, &sB[buf][r][ko]);
#else
            *(uint4*)&sA[buf][r][ko] = *(const uint4*)ga;
            *(uint4*)&sB[buf][r][ko] = *(const uint4*)gb;
#endif
        }
    };

    load_tile(0, 0);
    int cur = 0;
    for (int kt = 0; kt < K; kt += 32, cur ^= 1) {
        if (kt + 32 < K) {
            load_tile(cur ^ 1, kt + 32);   // prefetch next tile into other buffer
            WAIT_ASYNC(4);                 // current tile's 4 async ops completed (in-order)
        } else {
            WAIT_ASYNC(0);
        }
        __syncthreads();

        bf16x16 af[2], bfr[4];
        #pragma unroll
        for (int tm = 0; tm < 2; tm++) {
            int r = wm * 32 + tm * 16 + lrow;
            bf16x8 lo = *(const bf16x8*)&sA[cur][r][8 * lhalf];
            bf16x8 hi = *(const bf16x8*)&sA[cur][r][16 + 8 * lhalf];
            af[tm] = __builtin_shufflevector(lo, hi, 0,1,2,3,4,5,6,7,8,9,10,11,12,13,14,15);
        }
        #pragma unroll
        for (int tn = 0; tn < 4; tn++) {
            int n = wn * 64 + tn * 16 + lrow;
            bf16x8 lo = *(const bf16x8*)&sB[cur][n][8 * lhalf];
            bf16x8 hi = *(const bf16x8*)&sB[cur][n][16 + 8 * lhalf];
            bfr[tn] = __builtin_shufflevector(lo, hi, 0,1,2,3,4,5,6,7,8,9,10,11,12,13,14,15);
        }
        #pragma unroll
        for (int tm = 0; tm < 2; tm++)
            #pragma unroll
            for (int tn = 0; tn < 4; tn++)
                acc[tm][tn] = __builtin_amdgcn_wmma_f32_16x16x32_bf16(
                    false, af[tm], false, bfr[tn], (short)0, acc[tm][tn], false, false);
        __syncthreads();
    }

    // epilogue: D layout — vgpr r: M=r+8*lhalf, N=lrow
    #pragma unroll
    for (int tm = 0; tm < 2; tm++)
        #pragma unroll
        for (int tn = 0; tn < 4; tn++) {
            int col = bn0 + wn * 64 + tn * 16 + lrow;
            #pragma unroll
            for (int r = 0; r < 8; r++) {
                int row = bm0 + wm * 32 + tm * 16 + r + 8 * lhalf;
                float v = acc[tm][tn][r];
                if (EPI == 1) v += resid[(size_t)row * N + col];
                C[(size_t)row * N + col] = v;
            }
        }
}

// ---------- dt / lam / a / dt_scalar ----------
__global__ __launch_bounds__(256) void k_postproj(const float* __restrict__ proj,
                                                  const float* __restrict__ A_log,
                                                  const float* __restrict__ dt_bias,
                                                  float* __restrict__ dt, float* __restrict__ lam,
                                                  float* __restrict__ ae, float* __restrict__ dts) {
    int row = blockIdx.x * 256 + threadIdx.x;
    if (row >= ROWS) return;
    const float* pr = proj + (size_t)row * NPAD;
    float s = 0.f;
    #pragma unroll
    for (int h = 0; h < NH; h++) {
        float d = softplusf(pr[4224 + h] + dt_bias[h]);
        float l = sigm(pr[4240 + h]);
        float a = __expf(-__expf(A_log[h]) * d);
        dt[row * NH + h] = d; lam[row * NH + h] = l; ae[row * NH + h] = a;
        s += d;
    }
    dts[row] = s * (1.0f / NH);
}

// ---------- phi = cumsum_L(dt_scalar * theta) ----------
__global__ __launch_bounds__(32) void k_cumsum(const float* __restrict__ dts,
                                               const float* __restrict__ theta,
                                               float* __restrict__ phi) {
    int b = blockIdx.x, j = threadIdx.x;
    float acc = 0.f;
    for (int t = 0; t < SEQ; t++) {
        int row = b * SEQ + t;
        acc = fmaf(dts[row], theta[(size_t)row * 32 + j], acc);
        phi[(size_t)row * 32 + j] = acc;
    }
}

// ---------- rmsnorm + bias + RoPE on B/C ----------
__global__ __launch_bounds__(64) void k_bc(const float* __restrict__ proj,
                                           const float* __restrict__ bs, const float* __restrict__ cs,
                                           const float* __restrict__ bb, const float* __restrict__ cb,
                                           const float* __restrict__ phi,
                                           float* __restrict__ Bo, float* __restrict__ Co) {
    int row = blockIdx.x, n = threadIdx.x;
    const float* pr = proj + (size_t)row * NPAD;
    float b0 = pr[4096 + n], c0 = pr[4160 + n];
    __shared__ float rb[64], rc[64], sBn[64], sCn[64];
    rb[n] = b0 * b0; rc[n] = c0 * c0; __syncthreads();
    for (int o = 32; o > 0; o >>= 1) { if (n < o) { rb[n] += rb[n + o]; rc[n] += rc[n + o]; } __syncthreads(); }
    float ib = rsqrtf(rb[0] * (1.0f / DS) + 1e-6f);
    float ic = rsqrtf(rc[0] * (1.0f / DS) + 1e-6f);
    sBn[n] = b0 * ib * bs[n] + bb[n];
    sCn[n] = c0 * ic * cs[n] + cb[n];
    __syncthreads();
    if (n < 32) {
        float p = phi[(size_t)row * 32 + n];
        float cc = cosf(p), sn = sinf(p);
        float xr = sBn[n], xi = sBn[n + 32];
        Bo[(size_t)row * DS + n]      = xr * cc - xi * sn;
        Bo[(size_t)row * DS + n + 32] = xr * sn + xi * cc;
    } else {
        int m = n - 32;
        float p = phi[(size_t)row * 32 + m];
        float cc = cosf(p), sn = sinf(p);
        float xr = sCn[m], xi = sCn[m + 32];
        Co[(size_t)row * DS + m]      = xr * cc - xi * sn;
        Co[(size_t)row * DS + m + 32] = xr * sn + xi * cc;
    }
}

// ---------- SSD trapezoidal scan, fused silu/gate/D, bf16 out ----------
__global__ __launch_bounds__(128) void k_scan(const float* __restrict__ proj,
                                              const float* __restrict__ Bss, const float* __restrict__ Css,
                                              const float* __restrict__ dt, const float* __restrict__ lam,
                                              const float* __restrict__ ae, const float* __restrict__ Dp,
                                              unsigned short* __restrict__ ybuf) {
    int b = blockIdx.x >> 4, h = blockIdx.x & 15;
    int p = threadIdx.x;
    __shared__ float Bb[2][64], Cb[2][64];
    if (p < 64) { Bb[0][p] = 0.f; Bb[1][p] = 0.f; }
    else        { Cb[0][p - 64] = 0.f; Cb[1][p - 64] = 0.f; }
    __syncthreads();
    float hst[64];
    #pragma unroll
    for (int n = 0; n < 64; n++) hst[n] = 0.f;
    float xprev = 0.f;
    float Dh = Dp[h];
    for (int t = 0; t < SEQ; t++) {
        int row = b * SEQ + t;
        int cur = t & 1, prv = cur ^ 1;
        if (p < 64) Bb[cur][p]      = Bss[(size_t)row * DS + p];
        else        Cb[cur][p - 64] = Css[(size_t)row * DS + (p - 64)];
        __syncthreads();
        const float* pr = proj + (size_t)row * NPAD;
        float x = siluf(pr[h * HD + p]);
        float z = pr[DI + h * HD + p];
        float a = ae[row * NH + h], d = dt[row * NH + h], l = lam[row * NH + h];
        float w1c = d * l * x;                  // * B_t[n]
        float w2c = d * (1.0f - l) * a * xprev; // * B_{t-1}[n]
        float y = 0.f;
        #pragma unroll
        for (int n = 0; n < 64; n++) {
            float hn = fmaf(a, hst[n], fmaf(w1c, Bb[cur][n], w2c * Bb[prv][n]));
            hst[n] = hn;
            y = fmaf(hn, Cb[cur][n], y);
        }
        float yv = (y + x * Dh) * siluf(z);
        ybuf[(size_t)row * DI + h * HD + p] = f2bf(yv);
        xprev = x;
        __syncthreads();
    }
}

// ---------- launcher ----------
extern "C" void kernel_launch(void* const* d_in, const int* in_sizes, int n_in,
                              void* d_out, int out_size, void* d_ws, size_t ws_size,
                              hipStream_t stream) {
    const float* u          = (const float*)d_in[0];
    const float* norm_scale = (const float*)d_in[1];
    const float* norm_bias  = (const float*)d_in[2];
    const float* in_proj_w  = (const float*)d_in[3];
    const float* A_log      = (const float*)d_in[4];
    const float* dt_bias    = (const float*)d_in[5];
    const float* bcB_scale  = (const float*)d_in[6];
    const float* bcC_scale  = (const float*)d_in[7];
    const float* B_bias     = (const float*)d_in[8];
    const float* C_bias     = (const float*)d_in[9];
    const float* theta_w    = (const float*)d_in[10];
    const float* Dvec       = (const float*)d_in[11];
    const float* out_proj_w = (const float*)d_in[12];

    char* ws = (char*)d_ws;
    size_t off = 0;
    auto carve = [&](size_t bytes) { void* p = ws + off; off += (bytes + 255) & ~(size_t)255; return p; };

    unsigned short* W1T   = (unsigned short*)carve((size_t)NPAD * DM * 2);  // [N x K] transposed
    unsigned short* W2T   = (unsigned short*)carve((size_t)DM * DI * 2);    // [N x K] transposed
    unsigned short* XN    = (unsigned short*)carve((size_t)ROWS * DM * 2);
    float*          PROJ  = (float*)carve((size_t)ROWS * NPAD * 4);
    float*          THETA = (float*)carve((size_t)ROWS * 32 * 4);
    float*          PHI   = (float*)carve((size_t)ROWS * 32 * 4);
    float*          DT    = (float*)carve((size_t)ROWS * NH * 4);
    float*          LAM   = (float*)carve((size_t)ROWS * NH * 4);
    float*          AE    = (float*)carve((size_t)ROWS * NH * 4);
    float*          DTS   = (float*)carve((size_t)ROWS * 4);
    float*          BSSM  = (float*)carve((size_t)ROWS * DS * 4);
    float*          CSSM  = (float*)carve((size_t)ROWS * DS * 4);
    unsigned short* YBUF  = (unsigned short*)carve((size_t)ROWS * DI * 2);

    // weights -> bf16, transposed to [N x K] (pad in_proj N to 4352 with zero rows)
    {
        long long t1 = (long long)NPAD * DM;
        k_cvt_t<<<(unsigned)((t1 + 255) / 256), 256, 0, stream>>>(in_proj_w, W1T, DM, PROJN, t1);
        long long t2 = (long long)DM * DI;
        k_cvt_t<<<(unsigned)((t2 + 255) / 256), 256, 0, stream>>>(out_proj_w, W2T, DI, DM, t2);
    }
    k_layernorm<<<ROWS, 256, 0, stream>>>(u, norm_scale, norm_bias, XN);
    k_theta<<<ROWS / 8, 256, 0, stream>>>(u, theta_w, THETA);

    // in_proj GEMM: (16384 x 1024) @ (1024 x 4352)
    k_gemm<0><<<dim3(NPAD / 128, ROWS / 128), 256, 0, stream>>>(XN, W1T, PROJ, nullptr, ROWS, NPAD, DM);

    k_postproj<<<ROWS / 256, 256, 0, stream>>>(PROJ, A_log, dt_bias, DT, LAM, AE, DTS);
    k_cumsum<<<BATCH, 32, 0, stream>>>(DTS, THETA, PHI);
    k_bc<<<ROWS, 64, 0, stream>>>(PROJ, bcB_scale, bcC_scale, B_bias, C_bias, PHI, BSSM, CSSM);
    k_scan<<<BATCH * NH, 128, 0, stream>>>(PROJ, BSSM, CSSM, DT, LAM, AE, Dvec, YBUF);

    // out_proj GEMM + residual: (16384 x 2048) @ (2048 x 1024) + u
    k_gemm<1><<<dim3(DM / 128, ROWS / 128), 256, 0, stream>>>(YBUF, W2T, (float*)d_out, u, ROWS, DM, DI);
}
